// MessagePassingLayer_25400436589083
// MI455X (gfx1250) — matrix operand order repaired and checked by reading, not compiled, not run on hardware
//
#include <hip/hip_runtime.h>
#include <hip/hip_fp16.h>

#define NNODES 100000
#define NEDGES 1600000
#define DFEAT  32
#define LN_EPS 1e-3f

typedef __attribute__((ext_vector_type(16))) _Float16 v16h;
typedef __attribute__((ext_vector_type(8)))  float    v8f;

// D = A(16x32 f16) * B(32x16 f16) + C(16x16 f32)
static __device__ __forceinline__ v8f wmma_f16(v16h a, v16h b, v8f c) {
  return __builtin_amdgcn_wmma_f32_16x16x32_f16(
      /*neg_a=*/false, a, /*neg_b=*/false, b,
      /*c_mod=*/(short)0, c, /*reuse_a=*/false, /*reuse_b=*/false);
}

// Build the 16-bit A fragment (16x32, MxK) from a row-major f32 matrix [*,32].
// Per ISA 7.12.2: lanes 0-15 hold row M=lane, K={0..7,16..23};
//                 lanes 16-31 hold row M=lane-16, K={8..15,24..31}.
static __device__ __forceinline__ v16h load_a_frag(const float* __restrict__ base,
                                                   int r0, int lane, float scale) {
  int m     = lane & 15;
  int kbase = (lane >> 4) * 8;       // 0 or 8
  const float* row = base + (size_t)(r0 + m) * DFEAT;
  v16h a;
#pragma unroll
  for (int i = 0; i < 8; ++i) {
    a[i]     = (_Float16)(row[kbase + i] * scale);
    a[i + 8] = (_Float16)(row[kbase + 16 + i] * scale);
  }
  return a;
}

// Build the 16-bit B fragment (32x16, KxN) from row-major f32 W[K][ldn].
// Lanes 0-15: N=lane+n0, K=krow0+{0..7,16..23}; lanes 16-31: N=lane-16+n0, K=krow0+{8..15,24..31}.
static __device__ __forceinline__ v16h load_b_frag(const float* __restrict__ W,
                                                   int ldn, int krow0, int n0, int lane) {
  int n  = (lane & 15) + n0;
  int kb = (lane < 16) ? 0 : 8;
  const float* col = W + (size_t)(krow0 + kb) * ldn + n;
  v16h b;
#pragma unroll
  for (int h = 0; h < 16; ++h)
    b[h] = (_Float16)col[(size_t)(h < 8 ? h : h + 8) * ldn];
  return b;
}

__global__ void mpnn_zero_kernel(float* __restrict__ p, int n) {
  int i = blockIdx.x * blockDim.x + threadIdx.x;
  if (i < n) p[i] = 0.0f;
}

// msgs = relu(neighbor_feats @ Wm + bm), one wave per 16-row tile.
__global__ void mpnn_msg_kernel(const float* __restrict__ nbr,
                                const float* __restrict__ Wm,
                                const float* __restrict__ bm,
                                float* __restrict__ msgs, int ntiles) {
  int lane = threadIdx.x & 31;
  int tile = blockIdx.x * (blockDim.x >> 5) + (threadIdx.x >> 5);
  if (tile >= ntiles) return;
  int r0 = tile * 16;

  v16h a  = load_a_frag(nbr, r0, lane, 1.0f);
  v16h b0 = load_b_frag(Wm, DFEAT, 0, 0, lane);
  v16h b1 = load_b_frag(Wm, DFEAT, 0, 16, lane);
  v8f c0 = {}, c1 = {};
  c0 = wmma_f16(a, b0, c0);
  c1 = wmma_f16(a, b1, c1);

  int   n     = lane & 15;
  float bias0 = bm[n], bias1 = bm[n + 16];
  int   rbase = (lane < 16) ? 0 : 8;
#pragma unroll
  for (int v = 0; v < 8; ++v) {
    float x0 = fmaxf(c0[v] + bias0, 0.0f);
    float x1 = fmaxf(c1[v] + bias1, 0.0f);
    size_t g = (size_t)(r0 + rbase + v) * DFEAT;
    msgs[g + n]      = x0;
    msgs[g + n + 16] = x1;
  }
}

// One wave per edge: lane = feature. msgs/agg are L2-resident (12.8 MB each).
__global__ void mpnn_edge_kernel(const float* __restrict__ msgs,
                                 const float* __restrict__ ew,
                                 const int* __restrict__ src,
                                 const int* __restrict__ dst,
                                 float* __restrict__ agg,
                                 float* __restrict__ deg, int E) {
  int lane = threadIdx.x & 31;
  int e = blockIdx.x * (blockDim.x >> 5) + (threadIdx.x >> 5);
  if (e >= E) return;
  int   s = src[e];
  int   d = dst[e];
  float w = ew[e];
  float val = msgs[(size_t)s * DFEAT + lane] * w;
  atomicAdd(&agg[(size_t)d * DFEAT + lane], val);
  if (lane == 0) atomicAdd(&deg[d], w);
}

// gate/update GEMMs (K=64 -> 2 accumulating WMMAs each), blend, LayerNorm.
__global__ void mpnn_update_kernel(const float* __restrict__ node,
                                   const float* __restrict__ agg,
                                   const float* __restrict__ deg,
                                   const float* __restrict__ Wg,
                                   const float* __restrict__ bg,
                                   const float* __restrict__ Wu,
                                   const float* __restrict__ bu,
                                   const float* __restrict__ gamma,
                                   const float* __restrict__ beta,
                                   float* __restrict__ out, int ntiles) {
  int lane = threadIdx.x & 31;
  int tile = blockIdx.x * (blockDim.x >> 5) + (threadIdx.x >> 5);
  if (tile >= ntiles) return;
  int r0 = tile * 16;
  int m  = lane & 15;

  // normalize agg by weighted degree of the A-fragment's row (row = r0+m)
  float inv = 1.0f / fmaxf(deg[r0 + m], 1.0f);
  v16h aAgg  = load_a_frag(agg,  r0, lane, inv);
  v16h aNode = load_a_frag(node, r0, lane, 1.0f);

  v8f g0 = {}, g1 = {}, u0 = {}, u1 = {};
  // cat = [agg | node]; Wg/Wu rows 0..31 multiply agg, rows 32..63 multiply node
  g0 = wmma_f16(aAgg,  load_b_frag(Wg, DFEAT,  0,  0, lane), g0);
  g0 = wmma_f16(aNode, load_b_frag(Wg, DFEAT, 32,  0, lane), g0);
  g1 = wmma_f16(aAgg,  load_b_frag(Wg, DFEAT,  0, 16, lane), g1);
  g1 = wmma_f16(aNode, load_b_frag(Wg, DFEAT, 32, 16, lane), g1);
  u0 = wmma_f16(aAgg,  load_b_frag(Wu, DFEAT,  0,  0, lane), u0);
  u0 = wmma_f16(aNode, load_b_frag(Wu, DFEAT, 32,  0, lane), u0);
  u1 = wmma_f16(aAgg,  load_b_frag(Wu, DFEAT,  0, 16, lane), u1);
  u1 = wmma_f16(aNode, load_b_frag(Wu, DFEAT, 32, 16, lane), u1);

  float bg0 = bg[m], bg1 = bg[m + 16];
  float bu0 = bu[m], bu1 = bu[m + 16];
  int   rbase = (lane < 16) ? 0 : 8;

  v8f o0, o1;
#pragma unroll
  for (int v = 0; v < 8; ++v) {
    size_t grow = (size_t)(r0 + rbase + v) * DFEAT;
    float n0 = node[grow + m];
    float n1 = node[grow + m + 16];
    float gate0 = 1.0f / (1.0f + __expf(-(g0[v] + bg0)));
    float gate1 = 1.0f / (1.0f + __expf(-(g1[v] + bg1)));
    float up0 = fmaxf(u0[v] + bu0, 0.0f);
    float up1 = fmaxf(u1[v] + bu1, 0.0f);
    o0[v] = gate0 * up0 + (1.0f - gate0) * n0;
    o1[v] = gate1 * up1 + (1.0f - gate1) * n1;
  }

  // LayerNorm over D=32 fully in-register: xor-shuffle masks 1..8 reduce within
  // each 16-lane half, which is exactly one C/D row group per VGPR slot.
  float gam0 = gamma[m], gam1 = gamma[m + 16];
  float bet0 = beta[m],  bet1 = beta[m + 16];
#pragma unroll
  for (int v = 0; v < 8; ++v) {
    float s = o0[v] + o1[v];
#pragma unroll
    for (int msk = 1; msk < 16; msk <<= 1) s += __shfl_xor(s, msk, 32);
    float mean = s * (1.0f / 32.0f);
    float d0 = o0[v] - mean, d1 = o1[v] - mean;
    float sq = d0 * d0 + d1 * d1;
#pragma unroll
    for (int msk = 1; msk < 16; msk <<= 1) sq += __shfl_xor(sq, msk, 32);
    float rstd = rsqrtf(sq * (1.0f / 32.0f) + LN_EPS);
    size_t grow = (size_t)(r0 + rbase + v) * DFEAT;
    out[grow + m]      = d0 * rstd * gam0 + bet0;
    out[grow + m + 16] = d1 * rstd * gam1 + bet1;
  }
}

extern "C" void kernel_launch(void* const* d_in, const int* in_sizes, int n_in,
                              void* d_out, int out_size, void* d_ws, size_t ws_size,
                              hipStream_t stream) {
  const float* node  = (const float*)d_in[0];
  const float* nbr   = (const float*)d_in[1];
  const float* ew    = (const float*)d_in[2];
  const float* Wm    = (const float*)d_in[3];
  const float* bm    = (const float*)d_in[4];
  const float* Wg    = (const float*)d_in[5];
  const float* bg    = (const float*)d_in[6];
  const float* Wu    = (const float*)d_in[7];
  const float* bu    = (const float*)d_in[8];
  const float* gamma = (const float*)d_in[9];
  const float* beta  = (const float*)d_in[10];
  const int*   src   = (const int*)d_in[11];
  const int*   dst   = (const int*)d_in[12];
  float*       out   = (float*)d_out;

  const int N = NNODES, E = NEDGES;
  float* msgs = (float*)d_ws;                    // N*32 f32 = 12.8 MB
  float* agg  = msgs + (size_t)N * DFEAT;        // N*32 f32 = 12.8 MB
  float* deg  = agg  + (size_t)N * DFEAT;        // N    f32 =  0.4 MB

  const int ntiles = N / 16;                     // 6250, exact
  const int zcount = N * DFEAT + N;              // agg + deg (contiguous)

  mpnn_zero_kernel<<<(zcount + 255) / 256, 256, 0, stream>>>(agg, zcount);
  mpnn_msg_kernel<<<(ntiles + 3) / 4, 128, 0, stream>>>(nbr, Wm, bm, msgs, ntiles);
  mpnn_edge_kernel<<<(E + 7) / 8, 256, 0, stream>>>(msgs, ew, src, dst, agg, deg, E);
  mpnn_update_kernel<<<(ntiles + 3) / 4, 128, 0, stream>>>(
      node, agg, deg, Wg, bg, Wu, bu, gamma, beta, out, ntiles);
}
